// BinaryLinear_30580167147833
// MI455X (gfx1250) — compile-verified, matrix operand order checked by reading, not run on hardware
//
#include <hip/hip_runtime.h>

// ---------------------------------------------------------------------------
// BinaryLinear: y[M,N] = x[M,K] @ sign(W[N,K])^T + bias[N]
//   M = 8192, N = 4096, K = 4096  (f32 in / f32 out)
//
// Compute-bound GEMM (~860 FLOP/byte) -> f16 WMMA path with exact-ish split:
//   x = x_hi(f16) + x_lo(f16 residual);  sign(W) is exact in f16.
//   acc(f32) += A_hi*B ; acc += A_lo*B   (v_wmma_f32_16x16x32_f16)
//
// Pipeline: double-buffered LDS (one barrier / K-step) + register prefetch
// of the next global tile overlapping the WMMA burst.
// ---------------------------------------------------------------------------

typedef __attribute__((ext_vector_type(4)))  _Float16 v4h;
typedef __attribute__((ext_vector_type(8)))  _Float16 v8h;
typedef __attribute__((ext_vector_type(16))) _Float16 v16h;
typedef __attribute__((ext_vector_type(8)))  float    v8f;
typedef __attribute__((ext_vector_type(4)))  float    f32x4;

#define M_DIM 8192
#define N_DIM 4096
#define K_DIM 4096

#define BM 128
#define BN 128
#define BK 32
#define LDSK (BK + 8)   // 40 halves (80B) row pitch -> conflict-free b128 reads

// sign(v) as exact f16 via bit ops: (-1)^s * 1.0, or 0.0 if v == 0
__device__ __forceinline__ _Float16 sgnh(float v) {
    unsigned u = __float_as_uint(v);
    unsigned short h = (unsigned short)(((u >> 16) & 0x8000u) | 0x3C00u);
    _Float16 r = __builtin_bit_cast(_Float16, h);
    return (v == 0.0f) ? (_Float16)0.0f : r;
}

__global__ __launch_bounds__(256)
void binlin_wmma_kernel(const float* __restrict__ X,
                        const float* __restrict__ W,
                        const float* __restrict__ bias,
                        float* __restrict__ Y)
{
    // double-buffered tiles: [buf][plane][row][col]
    //   plane 0: x_hi, plane 1: x_lo, plane 2: sign(W)
    __shared__ _Float16 lds[2][3][BM][LDSK];   // 2*3*128*40*2B = 60 KB

    const int tid  = threadIdx.x;
    const int lane = tid & 31;
    const int wave = tid >> 5;
    const int l16  = lane & 15;
    const int hlf  = lane >> 4;          // 0: lanes 0-15, 1: lanes 16-31

    const int mBlk = blockIdx.y * BM;
    const int nBlk = blockIdx.x * BN;

    // 8 waves: 2 (M) x 4 (N); each wave owns a 64x32 output subtile
    const int mWave = (wave & 1) * 64;
    const int nWave = (wave >> 1) * 32;

    v8f acc[4][2] = {};                  // 4 M-tiles x 2 N-tiles of 16x16 f32

    // staging coordinates for this thread (1024 float4 / tile, 4 per thread)
    int srow[4], scol[4];
    #pragma unroll
    for (int it = 0; it < 4; ++it) {
        const int idx = tid + it * 256;
        srow[it] = idx >> 3;
        scol[it] = (idx & 7) << 2;
    }

    // ---- prologue: prefetch k=0 tile into registers --------------------
    f32x4 xr[4], wr[4];
    #pragma unroll
    for (int it = 0; it < 4; ++it) {
        xr[it] = *(const f32x4*)(X + (size_t)(mBlk + srow[it]) * K_DIM + scol[it]);
        wr[it] = *(const f32x4*)(W + (size_t)(nBlk + srow[it]) * K_DIM + scol[it]);
    }

    const int kA = hlf * 8;              // fragment K-chunk base (ISA 7.12.2)
    int buf = 0;

    for (int k0 = 0; k0 < K_DIM; k0 += BK, buf ^= 1) {
        // ---- stage prefetched registers -> LDS (convert f32 -> f16) ----
        #pragma unroll
        for (int it = 0; it < 4; ++it) {
            const int r = srow[it], c = scol[it];
            _Float16 h0 = (_Float16)xr[it].x, h1 = (_Float16)xr[it].y;
            _Float16 h2 = (_Float16)xr[it].z, h3 = (_Float16)xr[it].w;
            v4h hi = { h0, h1, h2, h3 };
            v4h lo = { (_Float16)(xr[it].x - (float)h0),
                       (_Float16)(xr[it].y - (float)h1),
                       (_Float16)(xr[it].z - (float)h2),
                       (_Float16)(xr[it].w - (float)h3) };
            v4h sg = { sgnh(wr[it].x), sgnh(wr[it].y),
                       sgnh(wr[it].z), sgnh(wr[it].w) };
            *(v4h*)&lds[buf][0][r][c] = hi;
            *(v4h*)&lds[buf][1][r][c] = lo;
            *(v4h*)&lds[buf][2][r][c] = sg;
        }
        __syncthreads();   // producers done -> consumers may read buf

        // ---- prefetch NEXT k tile into registers (overlaps WMMA) -------
        const int kn = k0 + BK;
        if (kn < K_DIM) {
            #pragma unroll
            for (int it = 0; it < 4; ++it) {
                xr[it] = *(const f32x4*)(X + (size_t)(mBlk + srow[it]) * K_DIM + kn + scol[it]);
                wr[it] = *(const f32x4*)(W + (size_t)(nBlk + srow[it]) * K_DIM + kn + scol[it]);
            }
        }

        // ---- fragment loads + WMMA -------------------------------------
        // 16-bit A/B lane layout: lane<16 holds K {0..7,16..23},
        // lane>=16 holds K {8..15,24..31}; row/col = lane%16.
        v16h bfrag[2];
        #pragma unroll
        for (int nt = 0; nt < 2; ++nt) {
            const int n = nWave + nt * 16 + l16;
            v8h b0 = *(const v8h*)&lds[buf][2][n][kA];
            v8h b1 = *(const v8h*)&lds[buf][2][n][16 + kA];
            bfrag[nt] = __builtin_shufflevector(b0, b1,
                0,1,2,3,4,5,6,7,8,9,10,11,12,13,14,15);
        }

        #pragma unroll
        for (int mt = 0; mt < 4; ++mt) {
            const int m = mWave + mt * 16 + l16;
            v8h a0 = *(const v8h*)&lds[buf][0][m][kA];
            v8h a1 = *(const v8h*)&lds[buf][0][m][16 + kA];
            v8h c0 = *(const v8h*)&lds[buf][1][m][kA];
            v8h c1 = *(const v8h*)&lds[buf][1][m][16 + kA];
            v16h ahi = __builtin_shufflevector(a0, a1,
                0,1,2,3,4,5,6,7,8,9,10,11,12,13,14,15);
            v16h alo = __builtin_shufflevector(c0, c1,
                0,1,2,3,4,5,6,7,8,9,10,11,12,13,14,15);

            #pragma unroll
            for (int nt = 0; nt < 2; ++nt) {
                acc[mt][nt] = __builtin_amdgcn_wmma_f32_16x16x32_f16(
                    false, ahi, false, bfrag[nt], (short)0, acc[mt][nt], false, false);
                acc[mt][nt] = __builtin_amdgcn_wmma_f32_16x16x32_f16(
                    false, alo, false, bfrag[nt], (short)0, acc[mt][nt], false, false);
            }
        }
        // no second barrier: buf is only rewritten after the NEXT barrier,
        // and every wave's reads completed before it signals that barrier.
    }

    // ---- epilogue: C layout (VGPR r -> M = 8*hlf + r, N = lane%16) -----
    #pragma unroll
    for (int mt = 0; mt < 4; ++mt) {
        #pragma unroll
        for (int nt = 0; nt < 2; ++nt) {
            const int col = nBlk + nWave + nt * 16 + l16;
            const float bv = bias[col];
            #pragma unroll
            for (int r = 0; r < 8; ++r) {
                const int row = mBlk + mWave + mt * 16 + hlf * 8 + r;
                Y[(size_t)row * N_DIM + col] = acc[mt][nt][r] + bv;
            }
        }
    }
}

extern "C" void kernel_launch(void* const* d_in, const int* in_sizes, int n_in,
                              void* d_out, int out_size, void* d_ws, size_t ws_size,
                              hipStream_t stream) {
    (void)in_sizes; (void)n_in; (void)out_size; (void)d_ws; (void)ws_size;
    const float* x = (const float*)d_in[0];
    const float* w = (const float*)d_in[1];
    const float* b = (const float*)d_in[2];
    float* y = (float*)d_out;

    dim3 grid(N_DIM / BN, M_DIM / BM);   // (32, 64)
    binlin_wmma_kernel<<<grid, 256, 0, stream>>>(x, w, b, y);
}
